// Attention_11141145166056
// MI455X (gfx1250) — matrix operand order repaired and checked by reading, not compiled, not run on hardware
//
#include <hip/hip_runtime.h>

// ---------------- problem constants ----------------
#define B_    4
#define S_    2048
#define DIM_  2048
#define H_    32
#define KVH_  8
#define HD_   64
#define NTOK  (B_ * S_)                    // 8192 tokens
#define QKV_N (H_*HD_ + 2*KVH_*HD_)        // 3072 = 2048 q + 512 k + 512 v
#define Q_OFF 0
#define K_OFF (H_*HD_)                     // 2048
#define V_OFF (H_*HD_ + KVH_*HD_)          // 2560
#define SCALE_ 0.125f                      // 64^-0.5

typedef __attribute__((ext_vector_type(16))) __bf16 v16bf;
typedef __attribute__((ext_vector_type(8)))  float  v8f;
typedef __attribute__((ext_vector_type(4)))  int    v4i;

// CDNA5 async global->LDS path (ASYNCcnt-tracked). Probe round 2 confirmed
// both builtins exist; param0 is v4i in AS1 (global), LDS side is AS3.
#if defined(__has_builtin)
#if __has_builtin(__builtin_amdgcn_global_load_async_to_lds_b128) && \
    __has_builtin(__builtin_amdgcn_s_wait_asynccnt)
#define HAS_ASYNC_LDS 1
#endif
#endif
#ifndef HAS_ASYNC_LDS
#define HAS_ASYNC_LDS 0
#endif

#if HAS_ASYNC_LDS
typedef __attribute__((address_space(1))) v4i* gp4_t;
typedef __attribute__((address_space(3))) v4i* lp4_t;
#define ASYNC_B128(gsrc, ldst) \
  __builtin_amdgcn_global_load_async_to_lds_b128( \
      (gp4_t)(void*)(gsrc), (lp4_t)(ldst), 0, 0)
#endif

// ---------------- bf16 helpers (bit-level, RNE) ----------------
__device__ __forceinline__ unsigned short f2bf(float f) {
  union { float f; unsigned u; } a; a.f = f;
  unsigned r = a.u + 0x7FFFu + ((a.u >> 16) & 1u);
  return (unsigned short)(r >> 16);
}
__device__ __forceinline__ float bf2f(unsigned short h) {
  union { unsigned u; float f; } a; a.u = ((unsigned)h) << 16;
  return a.f;
}

// ---------------- fp32 -> bf16 conversion ----------------
__global__ void cvt_f32_bf16(const float* __restrict__ in,
                             unsigned short* __restrict__ out, long n) {
  long i = (long)blockIdx.x * blockDim.x + threadIdx.x;
  if (i < n) out[i] = f2bf(in[i]);
}

// ---------------- WMMA GEMM: C[M,N] = A[M,K] * B[N,K]^T ----------------
// A, B bf16 row-major; C fp32 or bf16. 256 threads = 8 waves (wave32).
// Macro tile 128x256, K-step 32, double-buffered LDS.
// Wave grid 2(M) x 4(N); each wave owns a 64x64 tile = 4x4 grid of
// 16x16 v_wmma_f32_16x16x32_bf16 -> 16 WMMA per K-step per wave.
template<int OUT_BF16>
__global__ __launch_bounds__(256)
void gemm_bt(const unsigned short* __restrict__ A,
             const unsigned short* __restrict__ Bm,
             void* __restrict__ Cv,
             int M, int N, int K) {
  __shared__ __align__(16) unsigned short As[2][128][40];  // [buf][m][k] 20 KB
  __shared__ __align__(16) unsigned short Bs[2][256][40];  // [buf][n][k] 40 KB

  const int tid  = threadIdx.x;
  const int lane = tid & 31;
  const int wid  = tid >> 5;
  const int wr   = wid >> 2;        // 0..1 : 64-row band
  const int wc   = wid & 3;         // 0..3 : 64-col band
  const int r16  = lane & 15;
  const int half = lane >> 4;

  const int m0 = blockIdx.y * 128;
  const int n0 = blockIdx.x * 256;

  // cooperative load mapping:
  //   A slab 128x32: 256 thr -> row=tid>>1, 16-elem seg=(tid&1)*16  (32 B/thr)
  //   B slab 256x32: 256 thr -> row=tid, full 32-elem row           (64 B/thr)
  const int arow = tid >> 1;
  const int acol = (tid & 1) * 16;
  const unsigned short* Ag = A  + (size_t)(m0 + arow) * K + acol;
  const unsigned short* Bg = Bm + (size_t)(n0 + tid) * K;

  const v8f vzero = {0.f,0.f,0.f,0.f,0.f,0.f,0.f,0.f};
  v8f acc[4][4];
  #pragma unroll
  for (int i = 0; i < 4; i++)
    #pragma unroll
    for (int j = 0; j < 4; j++) acc[i][j] = vzero;

  // ---- preload slab 0 into buffer 0 ----
  {
    uint4 a0 = *(const uint4*)(Ag);
    uint4 a1 = *(const uint4*)(Ag + 8);
    uint4 b0 = *(const uint4*)(Bg);
    uint4 b1 = *(const uint4*)(Bg + 8);
    uint4 b2 = *(const uint4*)(Bg + 16);
    uint4 b3 = *(const uint4*)(Bg + 24);
    *(uint4*)&As[0][arow][acol]     = a0;
    *(uint4*)&As[0][arow][acol + 8] = a1;
    *(uint4*)&Bs[0][tid][0]         = b0;
    *(uint4*)&Bs[0][tid][8]         = b1;
    *(uint4*)&Bs[0][tid][16]        = b2;
    *(uint4*)&Bs[0][tid][24]        = b3;
  }
  __syncthreads();

  const int nk = K >> 5;
  for (int ki = 0; ki < nk; ki++) {
    const int cur  = ki & 1;
    const int nxt  = cur ^ 1;
    const int k0n  = (ki + 1) << 5;
    const bool more = (ki + 1 < nk);

#if HAS_ASYNC_LDS
    // stage next slab directly into LDS via async global->LDS (ASYNCcnt);
    // issued before the WMMA stream so memory latency hides under compute
    if (more) {
      ASYNC_B128(Ag + k0n,      &As[nxt][arow][acol]);
      ASYNC_B128(Ag + k0n + 8,  &As[nxt][arow][acol + 8]);
      ASYNC_B128(Bg + k0n,      &Bs[nxt][tid][0]);
      ASYNC_B128(Bg + k0n + 8,  &Bs[nxt][tid][8]);
      ASYNC_B128(Bg + k0n + 16, &Bs[nxt][tid][16]);
      ASYNC_B128(Bg + k0n + 24, &Bs[nxt][tid][24]);
    }
#else
    // stage next slab through registers; loads issued before compute so the
    // global latency hides under the WMMA stream
    uint4 a0, a1, b0, b1, b2, b3;
    if (more) {
      a0 = *(const uint4*)(Ag + k0n);
      a1 = *(const uint4*)(Ag + k0n + 8);
      b0 = *(const uint4*)(Bg + k0n);
      b1 = *(const uint4*)(Bg + k0n + 8);
      b2 = *(const uint4*)(Bg + k0n + 16);
      b3 = *(const uint4*)(Bg + k0n + 24);
    }
#endif

    // ---- compute from buffer `cur` ----
    // B fragments: lane = column n (r16), per-lane K contiguous half*16..+15
    union { uint4 q[2]; v16bf v; } bf[4];
    #pragma unroll
    for (int nt = 0; nt < 4; nt++) {
      const unsigned short* bp = &Bs[cur][wc*64 + nt*16 + r16][half*16];
      bf[nt].q[0] = *(const uint4*)(bp);
      bf[nt].q[1] = *(const uint4*)(bp + 8);
    }
    // A fragments: lane<16 row r16 K{0..15}; lane>=16 K{16..31}
    #pragma unroll
    for (int mt = 0; mt < 4; mt++) {
      union { uint4 q[2]; v16bf v; } af;
      const unsigned short* ap = &As[cur][wr*64 + mt*16 + r16][0];
      af.q[0] = *(const uint4*)(ap + half*16);
      af.q[1] = *(const uint4*)(ap + half*16 + 8);
      #pragma unroll
      for (int nt = 0; nt < 4; nt++) {
        acc[mt][nt] = __builtin_amdgcn_wmma_f32_16x16x32_bf16(
            false, af.v, false, bf[nt].v, (short)0, acc[mt][nt], false, false);
      }
    }

#if HAS_ASYNC_LDS
    if (more) __builtin_amdgcn_s_wait_asynccnt(0);
#else
    if (more) {
      *(uint4*)&As[nxt][arow][acol]     = a0;
      *(uint4*)&As[nxt][arow][acol + 8] = a1;
      *(uint4*)&Bs[nxt][tid][0]         = b0;
      *(uint4*)&Bs[nxt][tid][8]         = b1;
      *(uint4*)&Bs[nxt][tid][16]        = b2;
      *(uint4*)&Bs[nxt][tid][24]        = b3;
    }
#endif
    __syncthreads();
  }

  // C/D layout: VGPR v -> (M = v + half*8, N = r16) within each 16x16 tile
  #pragma unroll
  for (int mt = 0; mt < 4; mt++) {
    #pragma unroll
    for (int nt = 0; nt < 4; nt++) {
      const int col = n0 + wc*64 + nt*16 + r16;
      #pragma unroll
      for (int v = 0; v < 8; v++) {
        const int row = m0 + wr*64 + mt*16 + half*8 + v;
        const float val = acc[mt][nt][v];
        if (OUT_BF16)
          ((unsigned short*)Cv)[(size_t)row * N + col] = f2bf(val);
        else
          ((float*)Cv)[(size_t)row * N + col] = val;
      }
    }
  }
}

// ---------------- per-token head attention (+RoPE) ----------------
// One wave per token; lane = query head h (0..31). k/v staged in LDS.
__global__ __launch_bounds__(256)
void attn_kernel(const unsigned short* __restrict__ qkv,
                 const float* __restrict__ fcos,   // [S, 32]
                 const float* __restrict__ fsin,   // [S, 32]
                 unsigned short* __restrict__ attn) {
  __shared__ float kl[8][KVH_ * HD_];   // 8 tokens x 512 f32
  __shared__ float vl[8][KVH_ * HD_];

  const int lane = threadIdx.x & 31;
  const int wid  = threadIdx.x >> 5;
  const int t    = blockIdx.x * 8 + wid;       // token id, < 8192
  const int s    = t & (S_ - 1);               // position within sequence
  const unsigned short* base = qkv + (size_t)t * QKV_N;

  // stage k with RoPE applied: 256 pairs -> 8 per lane
  #pragma unroll
  for (int i = 0; i < 8; i++) {
    const int pg   = lane * 8 + i;             // 0..255
    const int head = pg >> 5;
    const int p    = pg & 31;
    const float x0 = bf2f(base[K_OFF + head*HD_ + 2*p]);
    const float x1 = bf2f(base[K_OFF + head*HD_ + 2*p + 1]);
    const float c  = fcos[s*32 + p];
    const float sn = fsin[s*32 + p];
    kl[wid][head*HD_ + 2*p]     = x0*c - x1*sn;
    kl[wid][head*HD_ + 2*p + 1] = x0*sn + x1*c;
  }
  // stage v (no RoPE)
  #pragma unroll
  for (int i = 0; i < 16; i++) {
    const int j = lane * 16 + i;               // 0..511
    vl[wid][j] = bf2f(base[V_OFF + j]);
  }
  __syncthreads();

  // q for head = lane, RoPE in registers
  float q[HD_];
  #pragma unroll
  for (int p = 0; p < 32; p++) {
    const float x0 = bf2f(base[Q_OFF + lane*HD_ + 2*p]);
    const float x1 = bf2f(base[Q_OFF + lane*HD_ + 2*p + 1]);
    const float c  = fcos[s*32 + p];
    const float sn = fsin[s*32 + p];
    q[2*p]     = x0*c - x1*sn;
    q[2*p + 1] = x0*sn + x1*c;
  }

  // scores over 32 repeated kv-heads (e -> kv head e>>2), LDS reads broadcast
  float sc[H_];
  float mx = -1e30f;
  #pragma unroll 4
  for (int e = 0; e < H_; e++) {
    const float* kk = &kl[wid][(e >> 2) * HD_];
    float d = 0.f;
    #pragma unroll
    for (int j = 0; j < HD_; j++) d += q[j] * kk[j];
    d *= SCALE_;
    sc[e] = d;
    mx = fmaxf(mx, d);
  }
  float sum = 0.f;
  #pragma unroll
  for (int e = 0; e < H_; e++) { sc[e] = __expf(sc[e] - mx); sum += sc[e]; }
  const float inv = 1.f / sum;

  float out[HD_];
  #pragma unroll
  for (int j = 0; j < HD_; j++) out[j] = 0.f;
  #pragma unroll 4
  for (int e = 0; e < H_; e++) {
    const float w = sc[e] * inv;
    const float* vv = &vl[wid][(e >> 2) * HD_];
    #pragma unroll
    for (int j = 0; j < HD_; j++) out[j] += w * vv[j];
  }

  unsigned short* dst = attn + (size_t)t * (H_*HD_) + lane * HD_;
  #pragma unroll
  for (int j = 0; j < HD_; j++) dst[j] = f2bf(out[j]);
}

// ---------------- host-side launch ----------------
extern "C" void kernel_launch(void* const* d_in, const int* in_sizes, int n_in,
                              void* d_out, int out_size, void* d_ws, size_t ws_size,
                              hipStream_t stream) {
  (void)in_sizes; (void)n_in; (void)out_size; (void)ws_size;
  const float* x  = (const float*)d_in[0];
  const float* wq = (const float*)d_in[1];
  const float* wk = (const float*)d_in[2];
  const float* wv = (const float*)d_in[3];
  const float* wo = (const float*)d_in[4];
  const float* fc = (const float*)d_in[5];
  const float* fs = (const float*)d_in[6];

  char* ws = (char*)d_ws;
  size_t off = 0;
  auto take = [&](size_t bytes) -> void* {
    void* p = ws + off;
    off += (bytes + 255) & ~(size_t)255;
    return p;
  };
  unsigned short* xb   = (unsigned short*)take((size_t)NTOK * DIM_ * 2);   // 33.5 MB
  unsigned short* wqkv = (unsigned short*)take((size_t)QKV_N * DIM_ * 2);  // 12.6 MB
  unsigned short* wob  = (unsigned short*)take((size_t)DIM_ * DIM_ * 2);   //  8.4 MB
  unsigned short* qkv  = (unsigned short*)take((size_t)NTOK * QKV_N * 2);  // 50.3 MB
  unsigned short* attn = (unsigned short*)take((size_t)NTOK * DIM_ * 2);   // 33.5 MB

  const long nx = (long)NTOK * DIM_;
  cvt_f32_bf16<<<(nx + 255) / 256, 256, 0, stream>>>(x, xb, nx);
  cvt_f32_bf16<<<((long)DIM_*DIM_ + 255) / 256, 256, 0, stream>>>(wq, wqkv, (long)DIM_*DIM_);
  cvt_f32_bf16<<<((long)KVH_*HD_*DIM_ + 255) / 256, 256, 0, stream>>>(
      wk, wqkv + (size_t)DIM_*DIM_, (long)KVH_*HD_*DIM_);
  cvt_f32_bf16<<<((long)KVH_*HD_*DIM_ + 255) / 256, 256, 0, stream>>>(
      wv, wqkv + (size_t)DIM_*DIM_ + (size_t)KVH_*HD_*DIM_, (long)KVH_*HD_*DIM_);
  cvt_f32_bf16<<<((long)DIM_*DIM_ + 255) / 256, 256, 0, stream>>>(wo, wob, (long)DIM_*DIM_);

  // qkv[8192,3072] = xb[8192,2048] * wqkv[3072,2048]^T   (bf16 out)
  gemm_bt<1><<<dim3(QKV_N / 256, NTOK / 128), 256, 0, stream>>>(
      xb, wqkv, qkv, NTOK, QKV_N, DIM_);

  // per-token RoPE + head-softmax attention
  attn_kernel<<<NTOK / 8, 256, 0, stream>>>(qkv, fc, fs, attn);

  // out[8192,2048] = attn[8192,2048] * wob[2048,2048]^T  (fp32 out -> d_out)
  gemm_bt<0><<<dim3(DIM_ / 256, NTOK / 128), 256, 0, stream>>>(
      attn, wob, d_out, NTOK, DIM_, DIM_);
}
